// WindowAttention_26508538151573
// MI455X (gfx1250) — matrix operand order repaired and checked by reading, not compiled, not run on hardware
//
#include <hip/hip_runtime.h>
#include <hip/hip_bf16.h>
#include <math.h>

typedef _Float16 f16;
typedef __attribute__((ext_vector_type(4)))  _Float16 v4h;
typedef __attribute__((ext_vector_type(8)))  _Float16 v8h;
typedef __attribute__((ext_vector_type(16))) _Float16 v16h;
typedef __attribute__((ext_vector_type(8)))  float    v8f;

#define N_TOK   49
#define NPAD    64
#define HEADS   8
#define HD      32
#define DIM     256
#define NW      64

// LDS layout (bytes).
#define XS_STRIDE 264                 // f16 units per row of x tile (padded)
#define VT_STRIDE 64                  // f16 units per row of transposed V
#define PB_STRIDE 72                  // f16 units per row of P (per head)
#define SMEM_XS   0                   // f16[49*264]          = 25872 B
#define SMEM_VT   25872               // f16[256*64]          = 32768 B
#define SMEM_PB   58640               // f16[8*64*72]         = 73728 B
#define SMEM_BL   132368              // float[169*8]         =  5408 B
#define SMEM_CM   137776              // int2[49*49]          = 19208 B  (mask bits, rel idx)
#define SMEM_TOTAL 157056

static __device__ inline v16h cat8(v8h lo, v8h hi) {
    return __builtin_shufflevector(lo, hi, 0,1,2,3,4,5,6,7,8,9,10,11,12,13,14,15);
}

// A operand (16-bit 16x32): lane r holds row M=r; halves i=0..15 -> K=half*16+i.
// Two contiguous 16B chunks per lane.
static __device__ inline v16h load_a_frag(const f16* row, int halfId) {
    v8h c0 = *(const v8h*)(row + halfId * 16);
    v8h c1 = *(const v8h*)(row + halfId * 16 + 8);
    return cat8(c0, c1);
}

// B operand (16-bit 32x16): lane (r,half) holds halves i -> B[K=half*16+i][N=r].
// 16 contiguous halves per lane.
static __device__ inline v16h load_b_frag(const f16* p) {
    v8h c0 = *(const v8h*)(p);
    v8h c1 = *(const v8h*)(p + 8);
    return cat8(c0, c1);
}

__global__ __launch_bounds__(256)
void WindowAttention_26508538151573_kernel(const float* __restrict__ x,
                                           const float* __restrict__ bias_table,
                                           const float* __restrict__ mask,
                                           const int*   __restrict__ rel_index,
                                           float*       __restrict__ out) {
    extern __shared__ char smem[];
    f16*   xs = (f16*)(smem + SMEM_XS);   // x[b] as f16, row-major [tok][256]
    f16*   vT = (f16*)(smem + SMEM_VT);   // transposed: [ch 256][tok 64]
    f16*   pb = (f16*)(smem + SMEM_PB);   // unnormalized exp (f16) per head [64][72]
    float* bl = (float*)(smem + SMEM_BL); // bias_table copy (169x8)
    int2*  cm = (int2*)(smem + SMEM_CM);  // {mask bits, rel idx} per (i,j)

    const int b      = blockIdx.x;
    const int tid    = threadIdx.x;
    const int h      = tid >> 5;          // one wave per head
    const int lane   = tid & 31;
    const int r      = lane & 15;
    const int halfId = lane >> 4;

    // ---------------- stage 0: fill LDS ----------------
    // zero-pad key rows 49..63 of vT first (disjoint from the fill below, race-free)
    {
        f16* vcol = vT + tid * VT_STRIDE; // 256 threads == 256 channels
#pragma unroll
        for (int row = N_TOK; row < NPAD; ++row) vcol[row] = (f16)0.0f;
    }
    // x tile: vectorized b128 global loads, packed f32->f16 converts.
    const float* xb = x + (size_t)b * (N_TOK * DIM);
    const float4* xb4 = (const float4*)xb;
    for (int e = tid; e < (N_TOK * DIM) / 4; e += 256) {
        float4 f = xb4[e];
        int row = e >> 6;                 // (e*4)/256
        int col = (e & 63) << 2;          // (e*4)%256
        v4h cv; cv[0] = (f16)f.x; cv[1] = (f16)f.y; cv[2] = (f16)f.z; cv[3] = (f16)f.w;
        *(v4h*)(xs + row * XS_STRIDE + col) = cv;
        vT[(col + 0) * VT_STRIDE + row] = cv[0];
        vT[(col + 1) * VT_STRIDE + row] = cv[1];
        vT[(col + 2) * VT_STRIDE + row] = cv[2];
        vT[(col + 3) * VT_STRIDE + row] = cv[3];
    }
    {   // bias table, vectorized
        const float4* bt4 = (const float4*)bias_table;
        float4* bl4 = (float4*)bl;
        for (int e = tid; e < (169 * HEADS) / 4; e += 256) bl4[e] = bt4[e];
    }
    {   // fused {mask, rel_index} records -> single ds_load_b64 in the epilogue
        const float* mrow = mask + (size_t)(b & (NW - 1)) * (N_TOK * N_TOK);
        for (int e = tid; e < N_TOK * N_TOK; e += 256) {
            int2 rec; rec.x = __float_as_int(mrow[e]); rec.y = rel_index[e];
            cm[e] = rec;
        }
    }
    __syncthreads();

    // ------- stage 1+2 fused per M-row: S-row = Q_mt * K^T, then exp epilogue -------
    // Only one mt row of accumulators (32 VGPRs) is live at a time; the next row's
    // WMMAs co-execute with this row's VALU epilogue.
    const float SCL2E = 0.17677669529663687f * 1.4426950408889634f; // scale * log2(e)
    const float L2E   = 1.4426950408889634f;
    f16* phead = pb + h * (NPAD * PB_STRIDE);

    v16h bfr[4];
#pragma unroll
    for (int nt = 0; nt < 4; ++nt) {
        int key = nt * 16 + r; if (key > N_TOK - 1) key = N_TOK - 1;   // clamp; cols zeroed later
        // b[i] = K^T[k = half*16+i][n = r] = x[key][h*32 + half*16 + i]
        bfr[nt] = load_b_frag(xs + key * XS_STRIDE + h * HD + halfId * 16);
    }
#pragma unroll
    for (int mt = 0; mt < 4; ++mt) {
        int qrow = mt * 16 + r; if (qrow > N_TOK - 1) qrow = N_TOK - 1;
        v16h a = load_a_frag(xs + qrow * XS_STRIDE + h * HD, halfId);
        v8f accr[4] = {};
#pragma unroll
        for (int nt = 0; nt < 4; ++nt)
            accr[nt] = __builtin_amdgcn_wmma_f32_16x16x32_f16(
                false, a, false, bfr[nt], (short)0, accr[nt], false, false);

        // exp epilogue for this row block: eh = 2^(l*log2e - 14); constant cancels in
        // the final division. Raw v_exp_f32: inputs < -126 flush to exact 0 (masked cols).
#pragma unroll
        for (int v = 0; v < 8; ++v) {
            int i    = mt * 16 + v + 8 * halfId;       // query row (C/D layout)
            int ic   = (i < N_TOK) ? i : (N_TOK - 1); // padded rows: finite garbage, never stored
            int base = ic * N_TOK;
            f16* prow = phead + i * PB_STRIDE;
            int2 rec[4];
#pragma unroll
            for (int nt = 0; nt < 4; ++nt) {           // batch the 4 b64 gathers
                int j  = nt * 16 + r;
                int jc = (j < N_TOK) ? j : (N_TOK - 1);
                rec[nt] = cm[base + jc];
            }
            float add[4];
#pragma unroll
            for (int nt = 0; nt < 4; ++nt)             // batch the 4 dependent bias loads
                add[nt] = bl[rec[nt].y * HEADS + h] + __int_as_float(rec[nt].x);
            f16 eh[4];
#pragma unroll
            for (int nt = 0; nt < 4; ++nt) {           // batch exp/cvt (fills TRANS hazards)
                int j   = nt * 16 + r;
                float t = fmaf(add[nt], L2E, -14.0f);
                t       = fmaf(accr[nt][v], SCL2E, t);
                t       = (j < N_TOK) ? t : -1e30f;    // folds away for nt<3
                eh[nt]  = (f16)__builtin_amdgcn_exp2f(t);
            }
#pragma unroll
            for (int nt = 0; nt < 4; ++nt) prow[nt * 16 + r] = eh[nt];
        }
    }
    // No barrier: stage 3 reads only this wave's P region (same-wave DS ordering)
    // and vT, which was fenced by the stage-0 barrier.

    // ---- stage 3: O_unnorm = Eh * V and rowsum = Eh * ones, all on the matrix pipe ----
    v16h vfr[2][2];
#pragma unroll
    for (int kt = 0; kt < 2; ++kt)
#pragma unroll
        for (int no = 0; no < 2; ++no)
            // b[i] = V[key = kt*32 + half*16 + i][dim = no*16 + r] via transposed copy
            vfr[kt][no] = load_b_frag(vT + (h * HD + no * 16 + r) * VT_STRIDE
                                         + kt * 32 + halfId * 16);
    v16h onesf;
#pragma unroll
    for (int t = 0; t < 16; ++t) onesf[t] = (f16)1.0f;

    v8f oacc[4][2] = {};
    v8f sacc[4]    = {};
#pragma unroll
    for (int mt = 0; mt < 4; ++mt) {
#pragma unroll
        for (int kt = 0; kt < 2; ++kt) {
            v16h a = load_a_frag(phead + (mt * 16 + r) * PB_STRIDE + kt * 32, halfId);
#pragma unroll
            for (int no = 0; no < 2; ++no)
                oacc[mt][no] = __builtin_amdgcn_wmma_f32_16x16x32_f16(
                    false, a, false, vfr[kt][no], (short)0, oacc[mt][no], false, false);
            sacc[mt] = __builtin_amdgcn_wmma_f32_16x16x32_f16(
                    false, a, false, onesf, (short)0, sacc[mt], false, false);
        }
    }

    // -------- stage 4: normalize with v_rcp and store (fp32) -> (B_, N, DIM) --------
    float* ob = out + (size_t)b * (N_TOK * DIM);
#pragma unroll
    for (int mt = 0; mt < 4; ++mt)
#pragma unroll
        for (int v = 0; v < 8; ++v) {
            int i = mt * 16 + v + 8 * halfId;
            if (i < N_TOK) {
                float inv = __builtin_amdgcn_rcpf(sacc[mt][v]); // rowsum replicated per lane
#pragma unroll
                for (int no = 0; no < 2; ++no)
                    ob[i * DIM + h * HD + no * 16 + r] = oacc[mt][no][v] * inv;
            }
        }
}

extern "C" void kernel_launch(void* const* d_in, const int* in_sizes, int n_in,
                              void* d_out, int out_size, void* d_ws, size_t ws_size,
                              hipStream_t stream) {
    const float* x          = (const float*)d_in[0];
    const float* bias_table = (const float*)d_in[1];
    const float* mask       = (const float*)d_in[2];
    const int*   rel_index  = (const int*)d_in[3];
    float*       out        = (float*)d_out;

    const int B_ = in_sizes[0] / (N_TOK * DIM);   // 4096
    dim3 grid(B_), block(256);
    WindowAttention_26508538151573_kernel<<<grid, block, SMEM_TOTAL, stream>>>(
        x, bias_table, mask, rel_index, out);
}
